// CorrectError_88330297409769
// MI455X (gfx1250) — matrix operand order repaired and checked by reading, not compiled, not run on hardware
//
#include <hip/hip_runtime.h>

// ---------------------------------------------------------------------------
// Exact L2 kNN (Q=2048, N=100000, D=512, K=32) via split-bf16 WMMA GEMM.
// score(q,j) = ||m_j||^2 - 2 * dot(q, m_j)   (||q||^2 dropped: per-row const)
// dot computed as (qh+ql).(mh+ml) ~= qh.mh + qh.ml + ql.mh  (3x bf16 WMMA,
// ~2^-17 relative error -> preserves exact top-32 ordering).
//
// v2: 32-query macro tile (two A fragment sets share every B fragment) ->
// each B byte from L2 feeds 6 WMMAs (~32 FLOP/L2-byte, half the B traffic
// of the 16-query version). Column chunk reduced to 1024 so the score
// staging LDS stays at the known-good 128 KB.
// ---------------------------------------------------------------------------

#define QN     2048
#define DK     512
#define NMEM   100000
#define CHUNK  1024
#define NCHUNK 98                  // 98 * 1024 = 100352 >= 100000
#define NPAD   (NCHUNK * CHUNK)    // 100352
#define KSEL   32
#define QTILE  32                  // queries per workgroup

typedef __bf16 v16bf __attribute__((ext_vector_type(16)));
typedef __bf16 v8bf  __attribute__((ext_vector_type(8)));
typedef float  v8f   __attribute__((ext_vector_type(8)));

static __device__ __forceinline__ unsigned short bf16_trunc_bits(float f) {
    unsigned u = __builtin_bit_cast(unsigned, f);
    return (unsigned short)(u >> 16);
}
static __device__ __forceinline__ float bf16_bits_to_f32(unsigned short b) {
    unsigned u = ((unsigned)b) << 16;
    return __builtin_bit_cast(float, u);
}

// ---------------- Phase 1a: query f32 -> bf16 hi/lo split -------------------
__global__ void knn_convert_q(const float* __restrict__ q,
                              unsigned short* __restrict__ qhi,
                              unsigned short* __restrict__ qlo) {
    int i = blockIdx.x * blockDim.x + threadIdx.x;
    if (i < QN * DK) {
        float v = q[i];
        unsigned short hb = bf16_trunc_bits(v);
        float hv = bf16_bits_to_f32(hb);
        qhi[i] = hb;
        qlo[i] = bf16_trunc_bits(v - hv);
    }
}

// ---------------- Phase 1b: exact f32 row norms ||m_j||^2 -------------------
__global__ void knn_row_norms(const float* __restrict__ mem,
                              float* __restrict__ mm) {
    int wave = threadIdx.x >> 5, lane = threadIdx.x & 31;
    int j = blockIdx.x * 8 + wave;
    if (j >= NPAD) return;
    if (j < NMEM) {
        float s = 0.f;
        const float* row = mem + (size_t)j * DK;
        #pragma unroll
        for (int k = lane; k < DK; k += 32) { float v = row[k]; s += v * v; }
        #pragma unroll
        for (int off = 16; off; off >>= 1) s += __shfl_down(s, off, 32);
        if (lane == 0) mm[j] = s;
    } else if (lane == 0) {
        mm[j] = 1.0e30f;   // pad rows can never enter the top-k
    }
}

// ---------------- Phase 1c: transpose-convert memory to k-major bf16 --------
// Mt[k][j] = mem[j][k] (hi/lo split), zero-filled for j >= NMEM. Makes every
// WMMA B-fragment element (lane = N, one K row each) a contiguous 32B read over 16 lanes.
__global__ void knn_transpose_convert(const float* __restrict__ mem,
                                      unsigned short* __restrict__ mthi,
                                      unsigned short* __restrict__ mtlo) {
    __shared__ float tile[32][33];
    int tx = threadIdx.x, ty = threadIdx.y;        // (32, 8)
    int j0 = blockIdx.x * 32, k0 = blockIdx.y * 32;
    #pragma unroll
    for (int i = 0; i < 4; ++i) {
        int jj = j0 + ty + i * 8;
        int kk = k0 + tx;
        tile[ty + i * 8][tx] = (jj < NMEM) ? mem[(size_t)jj * DK + kk] : 0.f;
    }
    __syncthreads();
    #pragma unroll
    for (int i = 0; i < 4; ++i) {
        int k = k0 + ty + i * 8;
        int j = j0 + tx;
        float v = tile[tx][ty + i * 8];
        unsigned short hb = bf16_trunc_bits(v);
        float hv = bf16_bits_to_f32(hb);
        size_t idx = (size_t)k * NPAD + j;
        mthi[idx] = hb;
        mtlo[idx] = bf16_trunc_bits(v - hv);
    }
}

// ---------------- Phase 2: WMMA GEMM + per-chunk top-32 ---------------------
static __device__ __forceinline__ v16bf load_a_frag(const unsigned short* p) {
    v8bf a0 = *(const v8bf*)p;
    v8bf a1 = *(const v8bf*)(p + 16);
    return __builtin_shufflevector(a0, a1, 0,1,2,3,4,5,6,7,
                                   8,9,10,11,12,13,14,15);
}

static __device__ __forceinline__ v8f wmma3(v16bf Ahi, v16bf Alo,
                                            v16bf Bhi, v16bf Blo, v8f acc) {
    acc = __builtin_amdgcn_wmma_f32_16x16x32_bf16(false, Alo, false, Bhi,
                                                  (short)0, acc, false, false);
    acc = __builtin_amdgcn_wmma_f32_16x16x32_bf16(false, Ahi, false, Blo,
                                                  (short)0, acc, false, false);
    acc = __builtin_amdgcn_wmma_f32_16x16x32_bf16(false, Ahi, false, Bhi,
                                                  (short)0, acc, false, false);
    return acc;
}

__launch_bounds__(128, 1)
__global__ void knn_gemm_select(const unsigned short* __restrict__ qhi,
                                const unsigned short* __restrict__ qlo,
                                const unsigned short* __restrict__ mthi,
                                const unsigned short* __restrict__ mtlo,
                                const float* __restrict__ mm,
                                float* __restrict__ candS,
                                int* __restrict__ candI) {
    __shared__ float smem[QTILE * CHUNK];        // 128 KB score staging
    const int tid  = threadIdx.x;
    const int wave = tid >> 5, lane = tid & 31;
    const int lm   = lane & 15;                  // A row / C column
    const int hs   = lane >> 4;                  // lane half select
    const int q0   = blockIdx.y * QTILE;
    const int c0   = blockIdx.x * CHUNK;

    for (int g = 0; g < 4; ++g) {                // 4 groups of 64 cols / wave
        const int j0 = c0 + wave * 256 + g * 64;
        v8f acc0[4], acc1[4];
        #pragma unroll
        for (int t = 0; t < 4; ++t) { acc0[t] = (v8f){}; acc1[t] = (v8f){}; }

        for (int kk = 0; kk < DK; kk += 32) {
            // A fragments for query rows [q0, q0+16) and [q0+16, q0+32):
            // lanes 0-15 hold K={kk..kk+7, kk+16..kk+23} of row lm,
            // lanes 16-31 hold K={kk+8..kk+15, kk+24..kk+31}.
            const size_t r0 = (size_t)(q0 + lm) * DK + kk + 8 * hs;
            const size_t r1 = (size_t)(q0 + 16 + lm) * DK + kk + 8 * hs;
            v16bf Ahi0 = load_a_frag(qhi + r0);
            v16bf Alo0 = load_a_frag(qlo + r0);
            v16bf Ahi1 = load_a_frag(qhi + r1);
            v16bf Alo1 = load_a_frag(qlo + r1);
            size_t krow = (size_t)(kk + 8 * hs) * NPAD;
            #pragma unroll
            for (int t = 0; t < 4; ++t) {
                size_t off = krow + (size_t)(j0 + t * 16 + lm);
                v16bf Bhi, Blo;   // column N=lm, K in same order as A
                #pragma unroll
                for (int s = 0; s < 16; ++s) {
                    size_t o = off + (size_t)(s < 8 ? s : s + 8) * NPAD;
                    Bhi[s] = __builtin_bit_cast(__bf16, mthi[o]);
                    Blo[s] = __builtin_bit_cast(__bf16, mtlo[o]);
                }
                acc0[t] = wmma3(Ahi0, Alo0, Bhi, Blo, acc0[t]);
                acc1[t] = wmma3(Ahi1, Alo1, Bhi, Blo, acc1[t]);
            }
        }
        // C layout: lanes 0-15: VGPR r -> M=r, N=lane; lanes 16-31: M=8+r.
        #pragma unroll
        for (int t = 0; t < 4; ++t) {
            int j   = j0 + t * 16 + lm;
            int col = wave * 256 + g * 64 + t * 16 + lm;
            float mmv = mm[j];
            #pragma unroll
            for (int r = 0; r < 8; ++r) {
                int m = r + 8 * hs;
                smem[m * CHUNK + col]          = mmv - 2.0f * acc0[t][r];
                smem[(16 + m) * CHUNK + col]   = mmv - 2.0f * acc1[t][r];
            }
        }
    }
    __syncthreads();

    // --- selection: 4 threads per query, each scans 256 columns -> local 32
    const int q = tid >> 2, part = tid & 3, cb = part * 256;
    const float* row = &smem[q * CHUNK];
    float bs[KSEL]; int bj[KSEL];
    float wv = -1.0e38f; int wp = 0;
    #pragma unroll 4
    for (int i = 0; i < KSEL; ++i) {
        float s = row[cb + i];
        bs[i] = s; bj[i] = c0 + cb + i;
        if (s > wv) { wv = s; wp = i; }
    }
    for (int i = KSEL; i < 256; ++i) {
        float s = row[cb + i];
        if (s < wv) {
            bs[wp] = s; bj[wp] = c0 + cb + i;
            wv = -1.0e38f;
            #pragma unroll
            for (int u = 0; u < KSEL; ++u)
                if (bs[u] > wv) { wv = bs[u]; wp = u; }
        }
    }
    __syncthreads();                       // all score reads done

    // overlay candidate arrays on the (now dead) score LDS
    float* cs = smem;                      // 32*128 floats
    int*   ci = (int*)(smem + QTILE * 128);
    int cbase = (q * 4 + part) * KSEL;
    #pragma unroll 4
    for (int i = 0; i < KSEL; ++i) { cs[cbase + i] = bs[i]; ci[cbase + i] = bj[i]; }
    __syncthreads();

    // --- merge 4x32 -> 32 per query (one thread per query)
    if (tid < QTILE) {
        int base  = tid * 128;
        size_t gb = ((size_t)(q0 + tid) * NCHUNK + blockIdx.x) * KSEL;
        for (int r = 0; r < KSEL; ++r) {
            float mv = 3.0e38f; int mp = base;
            for (int i = 0; i < 128; ++i) {
                float v = cs[base + i];
                if (v < mv) { mv = v; mp = base + i; }
            }
            candS[gb + r] = mv;
            candI[gb + r] = ci[mp];
            cs[mp] = 3.0e38f;
        }
    }
}

// ---------------- Phase 3: merge 98*32 candidates per query -----------------
__global__ void knn_final(const float* __restrict__ candS,
                          const int* __restrict__ candI,
                          const float* __restrict__ truev,
                          float* __restrict__ out) {
    const int NC = NCHUNK * KSEL;          // 3136
    __shared__ float ls[NCHUNK * KSEL];
    __shared__ int   li[NCHUNK * KSEL];
    int q = blockIdx.x, lane = threadIdx.x;
    size_t gb = (size_t)q * NC;
    for (int i = lane; i < NC; i += 32) { ls[i] = candS[gb + i]; li[i] = candI[gb + i]; }
    __syncthreads();
    float sum = 0.f;
    for (int r = 0; r < KSEL; ++r) {
        float mv = 3.0e38f; int mp = -1;
        for (int i = lane; i < NC; i += 32) {
            float v = ls[i];
            if (v < mv) { mv = v; mp = i; }
        }
        #pragma unroll
        for (int off = 16; off; off >>= 1) {
            float ov = __shfl_xor(mv, off, 32);
            int   op = __shfl_xor(mp, off, 32);
            if (ov < mv || (ov == mv && op >= 0 && (mp < 0 || op < mp))) { mv = ov; mp = op; }
        }
        if (mp >= 0) {
            if (lane == 0) sum += truev[li[mp]];
            if (lane == (mp & 31)) ls[mp] = 3.0e38f;
        }
        __syncthreads();
    }
    if (lane == 0) out[q] = sum * (1.0f / (float)KSEL);
}

// ---------------------------------------------------------------------------
extern "C" void kernel_launch(void* const* d_in, const int* in_sizes, int n_in,
                              void* d_out, int out_size, void* d_ws, size_t ws_size,
                              hipStream_t stream) {
    const float* hq  = (const float*)d_in[0];   // [2048, 512]
    const float* mem = (const float*)d_in[1];   // [100000, 512]
    const float* tv  = (const float*)d_in[2];   // [100000]
    float* out = (float*)d_out;                 // [2048]

    // workspace layout (bytes), all offsets 256B aligned; total ~250 MB
    char* ws = (char*)d_ws;
    const size_t QB  = (size_t)QN * DK * 2;            // 2 MB each
    const size_t MMB = (size_t)NPAD * 4;               // 0.4 MB
    const size_t MTB = (size_t)DK * NPAD * 2;          // 98 MB each
    const size_t CSB = (size_t)QN * NCHUNK * KSEL * 4; // 25.7 MB each
    unsigned short* qhi  = (unsigned short*)(ws);
    unsigned short* qlo  = (unsigned short*)(ws + QB);
    float*          mmv  = (float*)(ws + 2 * QB);
    unsigned short* mthi = (unsigned short*)(ws + 2 * QB + MMB);
    unsigned short* mtlo = (unsigned short*)(ws + 2 * QB + MMB + MTB);
    float*          cS   = (float*)(ws + 2 * QB + MMB + 2 * MTB);
    int*            cI   = (int*)(ws + 2 * QB + MMB + 2 * MTB + CSB);

    knn_convert_q<<<(QN * DK + 255) / 256, 256, 0, stream>>>(hq, qhi, qlo);
    knn_row_norms<<<NPAD / 8, 256, 0, stream>>>(mem, mmv);
    knn_transpose_convert<<<dim3(NPAD / 32, DK / 32), dim3(32, 8), 0, stream>>>(
        mem, mthi, mtlo);
    knn_gemm_select<<<dim3(NCHUNK, QN / QTILE), 128, 0, stream>>>(
        qhi, qlo, mthi, mtlo, mmv, cS, cI);
    knn_final<<<QN, 32, 0, stream>>>(cS, cI, tv, out);
}
